// NSF_AR_42417097016689
// MI455X (gfx1250) — matrix operand order repaired and checked by reading, not compiled, not run on hardware
//
#include <hip/hip_runtime.h>
#include <hip/hip_bf16.h>

// ---------------------------------------------------------------------------
// NSF-AR forward for MI455X (gfx1250, wave32, WMMA).
// Conditioner MLPs run on v_wmma_f32_16x16x32_f16; spline math on VALU.
// ---------------------------------------------------------------------------

typedef __attribute__((ext_vector_type(16))) _Float16 v16h;
typedef __attribute__((ext_vector_type(8)))  _Float16 v8h;
typedef __attribute__((ext_vector_type(8)))  float    v8f;

#define WAVES 4
#define ROWS_PER_WAVE 32
#define ROWS_PER_BLOCK (WAVES * ROWS_PER_WAVE)

// fragment regions inside d_ws (units: halves; 512 halves = 1 fragment = 32 lanes x 16 halves)
#define FRAG_HALVES 512
#define W1_FRAG_BASE 0                      // 15 dims * 4 ntiles            = 60 frags
#define W2_FRAG_BASE (60 * FRAG_HALVES)     // 15 dims * 4 ntiles * 2 ktiles = 120 frags
#define W3_FRAG_BASE (180 * FRAG_HALVES)    // 15 dims * 2 ntiles * 2 ktiles = 60 frags
#define TOTAL_FRAGS 240

static __device__ __forceinline__ v8f wmma16(v16h a, v16h b, v8f c) {
  // D = A(16x32 f16) * B(32x16 f16) + C(16x16 f32)
  return __builtin_amdgcn_wmma_f32_16x16x32_f16(
      /*neg_a=*/false, a, /*neg_b=*/false, b,
      /*c_mod=*/(short)0, c, /*reuse_a=*/false, /*reuse_b=*/false);
}

// Branch-free tanh: prefer the CDNA5 hardware transcendental (v_tanh_f32).
#if __has_builtin(__builtin_amdgcn_tanhf)
static __device__ __forceinline__ float fast_tanh(float x) {
  return __builtin_amdgcn_tanhf(x);
}
#else
static __device__ __forceinline__ float fast_tanh(float x) {
  // tanh(x) = 1 - 2/(exp(2x)+1); saturates correctly for |x| large, no branches.
  float t = __expf(2.0f * x);
  return 1.0f - 2.0f / (t + 1.0f);
}
#endif

// Branch-free, numerically stable softplus: max(x,0) + log1p(exp(-|x|)).
static __device__ __forceinline__ float softplus_f(float v) {
  return fmaxf(v, 0.0f) + __logf(1.0f + __expf(-fabsf(v)));
}

// ---------------------------------------------------------------------------
// Prepack: f32 weights -> f16 B-fragments in CDNA5 WMMA B layout.
// B tile (K=32 x N=16): lane L holds column n = L%16, halves j=0..15 map to
// k = hi*8 + (j<8 ? j : 8+j), hi = L/16. Zero-pad out-of-range K (layer1) / N (layer3).
// ---------------------------------------------------------------------------
__global__ void nsf_prepack(const float* __restrict__ w1,
                            const float* __restrict__ w2,
                            const float* __restrict__ w3,
                            _Float16* __restrict__ frag) {
  int t = blockIdx.x * blockDim.x + threadIdx.x;
  if (t >= TOTAL_FRAGS * 32) return;
  int fid  = t >> 5;
  int lane = t & 31;
  int n_local = lane & 15;
  int kofs = (lane < 16) ? 0 : 8;
  _Float16* dst = frag + (size_t)fid * FRAG_HALVES + lane * 16;

  if (fid < 60) {               // w1: fid = im1*4 + nt   (K padded: only k<i valid)
    int im1 = fid >> 2, nt = fid & 3, i = im1 + 1;
    #pragma unroll
    for (int j = 0; j < 16; ++j) {
      int k = kofs + (j < 8 ? j : 8 + j);
      int n = nt * 16 + n_local;
      float v = (k < i) ? w1[(im1 * 15 + k) * 64 + n] : 0.0f;
      dst[j] = (_Float16)v;
    }
  } else if (fid < 180) {       // w2: f = (im1*4 + nt)*2 + kt
    int f = fid - 60;
    int kt = f & 1, nt = (f >> 1) & 3, im1 = f >> 3;
    #pragma unroll
    for (int j = 0; j < 16; ++j) {
      int k = kt * 32 + kofs + (j < 8 ? j : 8 + j);
      int n = nt * 16 + n_local;
      dst[j] = (_Float16)w2[(im1 * 64 + k) * 64 + n];
    }
  } else {                      // w3: f = (im1*2 + nt)*2 + kt  (N padded: n<23 valid)
    int f = fid - 180;
    int kt = f & 1, nt = (f >> 1) & 1, im1 = f >> 2;
    #pragma unroll
    for (int j = 0; j < 16; ++j) {
      int k = kt * 32 + kofs + (j < 8 ? j : 8 + j);
      int n = nt * 16 + n_local;
      float v = (n < 23) ? w3[(im1 * 64 + k) * 23 + n] : 0.0f;
      dst[j] = (_Float16)v;
    }
  }
}

// ---------------------------------------------------------------------------
// Spline helpers (faithful to reference math).
// ---------------------------------------------------------------------------
// p[0..7] raw params -> edges e[0..8] via: 2B*softmax(p) -> softmax again ->
// min-width mix -> cumsum -> affine [-B,B] with forced endpoints.
static __device__ __forceinline__ void make_edges(const float* pp, float* e) {
  float t[8];
  float mx = pp[0];
  #pragma unroll
  for (int j = 1; j < 8; ++j) mx = fmaxf(mx, pp[j]);
  float s = 0.0f;
  #pragma unroll
  for (int j = 0; j < 8; ++j) { t[j] = __expf(pp[j] - mx); s += t[j]; }
  float inv = 6.0f / s;   // 2*B * softmax
  #pragma unroll
  for (int j = 0; j < 8; ++j) t[j] *= inv;
  float mx2 = t[0];
  #pragma unroll
  for (int j = 1; j < 8; ++j) mx2 = fmaxf(mx2, t[j]);
  float s2 = 0.0f;
  #pragma unroll
  for (int j = 0; j < 8; ++j) { t[j] = __expf(t[j] - mx2); s2 += t[j]; }
  float inv2 = (1.0f - 8.0f * 1e-3f) / s2;
  float cum = 0.0f;
  e[0] = -3.0f;
  #pragma unroll
  for (int j = 0; j < 8; ++j) { cum += 1e-3f + t[j] * inv2; e[j + 1] = 6.0f * cum - 3.0f; }
  e[8] = 3.0f;
}

static __device__ __forceinline__ void rqs_elem(float xv, const float* p,
                                                float& zout, float& ldout) {
  float e[9], ce[9];
  make_edges(p, e);        // widths
  make_edges(p + 8, ce);   // heights
  float wbin[8], hbin[8];
  #pragma unroll
  for (int j = 0; j < 8; ++j) { wbin[j] = e[j + 1] - e[j]; hbin[j] = ce[j + 1] - ce[j]; }
  float d[9];
  d[0] = 1.0f; d[8] = 1.0f;  // MIND + softplus(DERIV_CONST) == 1 by construction
  #pragma unroll
  for (int j = 1; j < 8; ++j) d[j] = 1e-3f + softplus_f(softplus_f(p[15 + j]));

  float xc = fminf(fmaxf(xv, -3.0f), 3.0f);
  int bin = -1;
  #pragma unroll
  for (int j = 0; j < 9; ++j) {
    float edge = (j == 8) ? (e[8] + 1e-6f) : e[j];
    bin += (xc >= edge) ? 1 : 0;
  }
  bin = bin < 0 ? 0 : (bin > 7 ? 7 : bin);

  float in_cw = e[0], in_w = wbin[0], in_ch = ce[0], in_h = hbin[0];
  float in_d = d[0], in_dp1 = d[1];
  #pragma unroll
  for (int j = 1; j < 8; ++j) {
    bool sel = (bin == j);
    in_cw  = sel ? e[j]     : in_cw;
    in_w   = sel ? wbin[j]  : in_w;
    in_ch  = sel ? ce[j]    : in_ch;
    in_h   = sel ? hbin[j]  : in_h;
    in_d   = sel ? d[j]     : in_d;
    in_dp1 = sel ? d[j + 1] : in_dp1;
  }
  float in_delta = in_h / in_w;
  float th  = (xc - in_cw) / in_w;
  float t1  = th * (1.0f - th);
  float num = in_h * (in_delta * th * th + in_d * t1);
  float den = in_delta + (in_d + in_dp1 - 2.0f * in_delta) * t1;
  float o   = in_ch + num / den;
  float omt = 1.0f - th;
  float dn  = in_delta * in_delta * (in_dp1 * th * th + 2.0f * in_delta * t1 + in_d * omt * omt);
  float ld  = __logf(dn) - 2.0f * __logf(den);
  bool inside = (xv >= -3.0f) && (xv <= 3.0f);
  zout  = inside ? o  : xv;
  ldout = inside ? ld : 0.0f;
}

// Load A fragment (16x32 f16, CDNA5 A layout) from a 64-wide row-major LDS row.
// lane halves: j<8 -> k = base + hi*8 + j ; j>=8 -> k = base + 16 + hi*8 + (j-8)
static __device__ __forceinline__ v16h load_a_frag(const _Float16* rowp, int kbase, int hi) {
  int kofs = kbase + hi * 8;
  v8h lo = *(const v8h*)(rowp + kofs);
  v8h hh = *(const v8h*)(rowp + kofs + 16);
  v16h a;
  #pragma unroll
  for (int j = 0; j < 8; ++j) { a[j] = lo[j]; a[j + 8] = hh[j]; }
  return a;
}

// ---------------------------------------------------------------------------
// Main kernel: 4 waves/block, 32 rows/wave (two 16-row M-tiles).
// ---------------------------------------------------------------------------
__global__ __launch_bounds__(WAVES * 32) void nsf_main(
    const float* __restrict__ x, const float* __restrict__ init_param,
    const float* __restrict__ b1, const float* __restrict__ b2,
    const float* __restrict__ b3, const _Float16* __restrict__ frag,
    float* __restrict__ out_z, float* __restrict__ out_ld, int nrows) {
  __shared__ __align__(16) float    xbuf [WAVES][ROWS_PER_WAVE][16];
  __shared__ __align__(16) _Float16 h1buf[WAVES][ROWS_PER_WAVE][64];
  __shared__ __align__(16) _Float16 h2buf[WAVES][ROWS_PER_WAVE][64];
  __shared__ __align__(16) float    pbuf [WAVES][ROWS_PER_WAVE][24];

  const int lane = threadIdx.x & 31;
  const int wave = threadIdx.x >> 5;
  const int m16  = lane & 15;
  const int hi   = lane >> 4;  // lane group 0/1
  const long rowbase = ((long)blockIdx.x * WAVES + wave) * ROWS_PER_WAVE;

  // stage x tile (32 rows x 16 cols), clamped at the tail
  {
    long row = rowbase + lane;
    long rc  = row < nrows ? row : (long)(nrows - 1);
    const float4* src = (const float4*)(x + rc * 16);
    float4* dst = (float4*)&xbuf[wave][lane][0];
    dst[0] = src[0]; dst[1] = src[1]; dst[2] = src[2]; dst[3] = src[3];
  }
  __syncthreads();

  float ld_acc = 0.0f;

  // ---- dim 0: broadcast init_param spline ----
  {
    float z, ldc;
    rqs_elem(xbuf[wave][lane][0], init_param, z, ldc);
    long row = rowbase + lane;
    if (row < nrows) out_z[row * 16 + 0] = z;
    ld_acc += ldc;
  }

  #pragma unroll 1
  for (int i = 1; i < 16; ++i) {
    const int im1 = i - 1;

    // ---------------- layer 1: x[:, :i] @ w1 + b1, tanh -> h1 ----------------
    v16h a1[2];
    #pragma unroll
    for (int mt = 0; mt < 2; ++mt) {
      int m = mt * 16 + m16;
      v16h a;
      #pragma unroll
      for (int j = 0; j < 16; ++j) {
        int k = (j < 8) ? (hi * 8 + j) : (16 + hi * 8 + (j - 8));
        int kc = k & 15;
        float v = (k < i) ? xbuf[wave][m][kc] : 0.0f;
        a[j] = (_Float16)v;
      }
      a1[mt] = a;
    }
    {
      const _Float16* w1f = frag + W1_FRAG_BASE + (size_t)(im1 * 4) * FRAG_HALVES;
      #pragma unroll
      for (int nt = 0; nt < 4; ++nt) {
        v16h b = *(const v16h*)(w1f + (size_t)nt * FRAG_HALVES + lane * 16);
        float bias = b1[im1 * 64 + nt * 16 + m16];
        #pragma unroll
        for (int mt = 0; mt < 2; ++mt) {
          v8f acc = {0.f, 0.f, 0.f, 0.f, 0.f, 0.f, 0.f, 0.f};
          acc = wmma16(a1[mt], b, acc);
          #pragma unroll
          for (int r = 0; r < 8; ++r) {
            int m = mt * 16 + r + hi * 8;     // D layout: vgpr r -> M=r (+8 for hi lanes)
            h1buf[wave][m][nt * 16 + m16] = (_Float16)fast_tanh(acc[r] + bias);
          }
        }
      }
    }
    __syncthreads();

    // ---------------- layer 2: h1 @ w2 + b2, tanh -> h2 ----------------
    v16h a2[2][2];
    #pragma unroll
    for (int mt = 0; mt < 2; ++mt)
      #pragma unroll
      for (int kt = 0; kt < 2; ++kt)
        a2[mt][kt] = load_a_frag(&h1buf[wave][mt * 16 + m16][0], kt * 32, hi);
    {
      const _Float16* w2f = frag + W2_FRAG_BASE;
      #pragma unroll
      for (int nt = 0; nt < 4; ++nt) {
        v16h bk0 = *(const v16h*)(w2f + (size_t)((im1 * 4 + nt) * 2 + 0) * FRAG_HALVES + lane * 16);
        v16h bk1 = *(const v16h*)(w2f + (size_t)((im1 * 4 + nt) * 2 + 1) * FRAG_HALVES + lane * 16);
        float bias = b2[im1 * 64 + nt * 16 + m16];
        #pragma unroll
        for (int mt = 0; mt < 2; ++mt) {
          v8f acc = {0.f, 0.f, 0.f, 0.f, 0.f, 0.f, 0.f, 0.f};
          acc = wmma16(a2[mt][0], bk0, acc);
          acc = wmma16(a2[mt][1], bk1, acc);
          #pragma unroll
          for (int r = 0; r < 8; ++r) {
            int m = mt * 16 + r + hi * 8;
            h2buf[wave][m][nt * 16 + m16] = (_Float16)fast_tanh(acc[r] + bias);
          }
        }
      }
    }
    __syncthreads();

    // ---------------- layer 3: h2 @ w3 + b3 -> params (16+7 cols) ----------------
    v16h a3[2][2];
    #pragma unroll
    for (int mt = 0; mt < 2; ++mt)
      #pragma unroll
      for (int kt = 0; kt < 2; ++kt)
        a3[mt][kt] = load_a_frag(&h2buf[wave][mt * 16 + m16][0], kt * 32, hi);
    {
      const _Float16* w3f = frag + W3_FRAG_BASE;
      #pragma unroll
      for (int nt = 0; nt < 2; ++nt) {
        v16h bk0 = *(const v16h*)(w3f + (size_t)((im1 * 2 + nt) * 2 + 0) * FRAG_HALVES + lane * 16);
        v16h bk1 = *(const v16h*)(w3f + (size_t)((im1 * 2 + nt) * 2 + 1) * FRAG_HALVES + lane * 16);
        int n = nt * 16 + m16;
        float bias = (n < 23) ? b3[im1 * 23 + n] : 0.0f;
        #pragma unroll
        for (int mt = 0; mt < 2; ++mt) {
          v8f acc = {0.f, 0.f, 0.f, 0.f, 0.f, 0.f, 0.f, 0.f};
          acc = wmma16(a3[mt][0], bk0, acc);
          acc = wmma16(a3[mt][1], bk1, acc);
          if (n < 23) {
            #pragma unroll
            for (int r = 0; r < 8; ++r) {
              int m = mt * 16 + r + hi * 8;
              pbuf[wave][m][n] = acc[r] + bias;
            }
          }
        }
      }
    }
    __syncthreads();

    // ---------------- spline: one row per lane ----------------
    {
      float z, ldc;
      rqs_elem(xbuf[wave][lane][i], &pbuf[wave][lane][0], z, ldc);
      long row = rowbase + lane;
      if (row < nrows) out_z[row * 16 + i] = z;
      ld_acc += ldc;
    }
  }

  long row = rowbase + lane;
  if (row < nrows) out_ld[row] = ld_acc;
}

// ---------------------------------------------------------------------------
extern "C" void kernel_launch(void* const* d_in, const int* in_sizes, int n_in,
                              void* d_out, int out_size, void* d_ws, size_t ws_size,
                              hipStream_t stream) {
  const float* x          = (const float*)d_in[0];
  const float* init_param = (const float*)d_in[1];
  const float* w1         = (const float*)d_in[2];
  const float* b1         = (const float*)d_in[3];
  const float* w2         = (const float*)d_in[4];
  const float* b2         = (const float*)d_in[5];
  const float* w3         = (const float*)d_in[6];
  const float* b3         = (const float*)d_in[7];
  const int nrows = in_sizes[0] / 16;

  _Float16* frag = (_Float16*)d_ws;  // needs 240 * 512 * 2 = 245,760 bytes

  nsf_prepack<<<(TOTAL_FRAGS * 32 + 255) / 256, 256, 0, stream>>>(w1, w2, w3, frag);

  float* out_z  = (float*)d_out;
  float* out_ld = out_z + (size_t)nrows * 16;
  const int nblocks = (nrows + ROWS_PER_BLOCK - 1) / ROWS_PER_BLOCK;
  nsf_main<<<nblocks, WAVES * 32, 0, stream>>>(x, init_param, b1, b2, b3, frag,
                                               out_z, out_ld, nrows);
}
